// MultiHeadSelfAttention_76467597738343
// MI455X (gfx1250) — matrix operand order repaired and checked by reading, not compiled, not run on hardware
//
#include <hip/hip_runtime.h>
#include <stdint.h>

// MHSA: B=4, N=2048, E=768, H=12, HD=64.  Compute-bound (~90 GFLOP vs ~150MB
// HBM traffic) -> f16 WMMA (v_wmma_f32_16x16x32_f16) with fp32 accumulation,
// flash-attention streaming softmax so NxN scores never hit HBM.

#define BQ   4
#define SEQ  2048
#define EMB  768
#define NH   12
#define HD   64
#define QKVN 2304   // 3*EMB
#define SCALER 64.0f
#define L2E 1.44269504f

#define ASG __attribute__((address_space(1)))   // global
#define ASL __attribute__((address_space(3)))   // LDS

#if __has_builtin(__builtin_amdgcn_global_load_async_to_lds_b128)
#define ASYNC_LDS 1
#else
#define ASYNC_LDS 0
#endif

typedef __attribute__((ext_vector_type(16))) _Float16 v16h;
typedef __attribute__((ext_vector_type(8)))  float    v8f;
typedef __attribute__((__vector_size__(4 * sizeof(int)))) int v4i;

union FragU { v16h h; uint32_t u[8]; };

// ---- WMMA D = A(16x32) * B(32x16) + C, f16 in / f32 accum -------------------
__device__ __forceinline__ v8f wmma32(v16h a, v16h b, v8f c) {
  return __builtin_amdgcn_wmma_f32_16x16x32_f16(
      /*neg_a=*/false, a, /*neg_b=*/false, b,
      /*c_mod=*/(short)0, c, /*reuse_a=*/false, /*reuse_b=*/false);
}

// A-fragment loader: s points at a row-major 16xK f16 tile (leading dim ld,
// halves).  Lane layout per ISA: lanes 0-15 row=lane, K in {0..7,16..23};
// lanes 16-31 same rows, K in {8..15,24..31}.  ld must be even.
__device__ __forceinline__ v16h frag_a(const _Float16* s, int ld, int lane) {
  const int row = lane & 15;
  const int kb2 = (lane >> 4) << 2;            // half-offset/2: 0 or 4 dwords
  const uint32_t* p = (const uint32_t*)(s + row * ld);
  FragU f;
#pragma unroll
  for (int j = 0; j < 4; ++j) f.u[j]     = p[kb2 + j];        // K = kb+2j,+1
#pragma unroll
  for (int j = 0; j < 4; ++j) f.u[4 + j] = p[8 + kb2 + j];    // K = 16+kb+2j
  return f.h;
}

// B-fragment loader from a tile stored column-major-in-K: element (k,n) at
// s[n*ld + k].  Lanes 0-15: K in {0..7,16..23} at col=lane; lanes 16-31:
// K in {8..15,24..31}.  ld must be even.
__device__ __forceinline__ v16h frag_bt(const _Float16* s, int ld, int lane) {
  const int col = lane & 15;
  const int kb2 = (lane >> 4) << 2;            // 0 or 4 dwords
  const uint32_t* p = (const uint32_t*)(s + col * ld);
  FragU f;
#pragma unroll
  for (int j = 0; j < 4; ++j) f.u[j]     = p[kb2 + j];
#pragma unroll
  for (int j = 0; j < 4; ++j) f.u[4 + j] = p[8 + kb2 + j];
  return f.h;
}

// ============================================================================
// Kernel 1: QKV = x @ w_qkv, scattered to Q/K/V f16 buffers [B,H,N,HD].
// Tile 128(M) x 64(N), 8 waves (4x2), each wave 32x32 = 2x2 WMMA tiles.
// Staging uses 128-bit global loads (float4) -> f16 packed LDS stores.
// ============================================================================
__global__ __launch_bounds__(256) void qkv_gemm_kernel(
    const float* __restrict__ x, const float* __restrict__ w,
    _Float16* __restrict__ qb, _Float16* __restrict__ kb,
    _Float16* __restrict__ vb) {
  __shared__ _Float16 As[128 * 40];   // padded ld=40
  __shared__ _Float16 Bt[64 * 40];    // transposed: [n][k]
  const int t    = threadIdx.x;
  const int lane = t & 31;
  const int wid  = t >> 5;
  const int wm   = wid & 3, wn = wid >> 2;
  const int mBase = blockIdx.y * 128;
  const int nBase = blockIdx.x * 64;

  v8f acc[2][2];
#pragma unroll
  for (int i = 0; i < 2; ++i)
#pragma unroll
    for (int j = 0; j < 2; ++j) acc[i][j] = {};

  for (int k0 = 0; k0 < EMB; k0 += 32) {
    __syncthreads();
#pragma unroll
    for (int i = 0; i < 4; ++i) {          // A: 128x32 fp32 = 1024 float4
      int idx = t + i * 256;
      int r = idx >> 3, c4 = (idx & 7) * 4;
      float4 f = *(const float4*)&x[(size_t)(mBase + r) * EMB + k0 + c4];
      union { _Float16 h[4]; uint64_t u; } pk;
      pk.h[0] = (_Float16)f.x; pk.h[1] = (_Float16)f.y;
      pk.h[2] = (_Float16)f.z; pk.h[3] = (_Float16)f.w;
      *(uint64_t*)&As[r * 40 + c4] = pk.u;
    }
#pragma unroll
    for (int i = 0; i < 2; ++i) {          // B: 32x64 fp32 = 512 float4
      int idx = t + i * 256;
      int k = idx >> 4, n4 = (idx & 15) * 4;
      float4 f = *(const float4*)&w[(size_t)(k0 + k) * QKVN + nBase + n4];
      Bt[(n4 + 0) * 40 + k] = (_Float16)f.x;
      Bt[(n4 + 1) * 40 + k] = (_Float16)f.y;
      Bt[(n4 + 2) * 40 + k] = (_Float16)f.z;
      Bt[(n4 + 3) * 40 + k] = (_Float16)f.w;
    }
    if (k0 + 32 < EMB) {                   // warm L2 for next k-slice
      __builtin_prefetch(&x[(size_t)(mBase + (t >> 1)) * EMB + k0 + 32], 0, 1);
      if (t < 32)
        __builtin_prefetch(&w[(size_t)(k0 + 32 + t) * QKVN + nBase], 0, 1);
    }
    __syncthreads();

    v16h a0 = frag_a(As + (wm * 32)      * 40, 40, lane);
    v16h a1 = frag_a(As + (wm * 32 + 16) * 40, 40, lane);
    v16h b0 = frag_bt(Bt + (wn * 32)      * 40, 40, lane);
    v16h b1 = frag_bt(Bt + (wn * 32 + 16) * 40, 40, lane);
    acc[0][0] = wmma32(a0, b0, acc[0][0]);
    acc[0][1] = wmma32(a0, b1, acc[0][1]);
    acc[1][0] = wmma32(a1, b0, acc[1][0]);
    acc[1][1] = wmma32(a1, b1, acc[1][1]);
  }

  // Scatter epilogue: col -> (q/k/v, head, dim); row -> (batch, seq)
  const int rofs = (lane >> 4) << 3;
  const int cofs = lane & 15;
#pragma unroll
  for (int i = 0; i < 2; ++i)
#pragma unroll
    for (int j = 0; j < 2; ++j)
#pragma unroll
      for (int r = 0; r < 8; ++r) {
        int grow = mBase + wm * 32 + i * 16 + r + rofs;
        int gcol = nBase + wn * 32 + j * 16 + cofs;
        int b = grow >> 11, n = grow & (SEQ - 1);
        int which = gcol / EMB;
        int rem = gcol - which * EMB;
        int h = rem >> 6, d = rem & 63;
        _Float16* dst = (which == 0) ? qb : (which == 1) ? kb : vb;
        dst[(((size_t)b * NH + h) * SEQ + n) * HD + d] = (_Float16)acc[i][j][r];
      }
}

// ============================================================================
// Kernel 2: flash attention per (64 q-rows, b*H).  4 waves, 16 q-rows/wave.
// S = (Q K^T) * 64, online softmax, O = P V.  K chunk staged with CDNA5
// async-to-LDS (ASYNCcnt) when available; V transposed through VGPRs.
// ============================================================================
__global__ __launch_bounds__(128) void attn_kernel(
    const _Float16* __restrict__ qb, const _Float16* __restrict__ kb,
    const _Float16* __restrict__ vb, _Float16* __restrict__ att) {
  __shared__ _Float16 Kc[64 * 80];      // [key][dim], padded ld=80
  __shared__ _Float16 Vt[64 * 80];      // [dim][key], padded ld=80
  __shared__ _Float16 Pl[4][16 * 72];   // per-wave P scratch, ld=72
  const int t = threadIdx.x, lane = t & 31, wid = t >> 5;
  const int bh = blockIdx.y;
  const int q0 = blockIdx.x * 64;
  const _Float16* Q = qb + (size_t)bh * SEQ * HD;
  const _Float16* K = kb + (size_t)bh * SEQ * HD;
  const _Float16* V = vb + (size_t)bh * SEQ * HD;

  // Q fragments for this wave's 16 rows (2 slices over the 64-d contraction)
  v16h aQ0 = frag_a(Q + (size_t)(q0 + wid * 16) * HD,      HD, lane);
  v16h aQ1 = frag_a(Q + (size_t)(q0 + wid * 16) * HD + 32, HD, lane);

  float rm[8], rl[8];
  v8f o[4];
#pragma unroll
  for (int r = 0; r < 8; ++r) { rm[r] = -3.0e38f; rl[r] = 0.0f; }
#pragma unroll
  for (int j = 0; j < 4; ++j) o[j] = {};

  for (int c0 = 0; c0 < SEQ; c0 += 64) {
    __syncthreads();
    // Stage 64x64 K (row-major, async direct-to-LDS) and V (transposed)
#pragma unroll
    for (int i = 0; i < 4; ++i) {
      int idx = t + i * 128;                       // 512 x 16B
      int key = idx >> 3, d8 = (idx & 7) * 8;
#if ASYNC_LDS
      __builtin_amdgcn_global_load_async_to_lds_b128(
          (ASG v4i*)(uintptr_t)(K + (size_t)(c0 + key) * HD + d8),
          (ASL v4i*)(uint32_t)(uintptr_t)&Kc[key * 80 + d8], 0, 0);
#else
      uint4 kk = *(const uint4*)(K + (size_t)(c0 + key) * HD + d8);
      *(uint4*)&Kc[key * 80 + d8] = kk;
#endif
      uint4 vv = *(const uint4*)(V + (size_t)(c0 + key) * HD + d8);
      const _Float16* vh = (const _Float16*)&vv;
#pragma unroll
      for (int j = 0; j < 8; ++j) Vt[(d8 + j) * 80 + key] = vh[j];
    }
    if (c0 + 64 < SEQ) {                           // warm L2 for next chunk
      __builtin_prefetch(K + (size_t)(c0 + 64 + (t >> 1)) * HD, 0, 1);
      __builtin_prefetch(V + (size_t)(c0 + 64 + (t >> 1)) * HD, 0, 1);
    }
#if ASYNC_LDS
#if __has_builtin(__builtin_amdgcn_s_wait_asynccnt)
    __builtin_amdgcn_s_wait_asynccnt(0);
#else
    asm volatile("s_wait_asynccnt 0x0" ::: "memory");
#endif
#endif
    __syncthreads();

    // S = Q K^T  (per wave: 16 x 64 scores)
    v8f sc[4];
#pragma unroll
    for (int j = 0; j < 4; ++j) sc[j] = {};
#pragma unroll
    for (int j = 0; j < 4; ++j) {
      v16h b0 = frag_bt(Kc + (j * 16) * 80,      80, lane);  // dims 0..31
      v16h b1 = frag_bt(Kc + (j * 16) * 80 + 32, 80, lane);  // dims 32..63
      sc[j] = wmma32(aQ0, b0, sc[j]);
      sc[j] = wmma32(aQ1, b1, sc[j]);
    }
#pragma unroll
    for (int j = 0; j < 4; ++j)
#pragma unroll
      for (int r = 0; r < 8; ++r) sc[j][r] *= SCALER;

    // Online softmax.  C-frag: vgpr r holds row r (lanes 0-15) / row r+8
    // (lanes 16-31); 16-lane xor butterflies reduce both halves at once.
    float mnew[8], corr[8];
#pragma unroll
    for (int r = 0; r < 8; ++r) {
      float mx = fmaxf(fmaxf(sc[0][r], sc[1][r]), fmaxf(sc[2][r], sc[3][r]));
      mx = fmaxf(mx, __shfl_xor(mx, 1, 32));
      mx = fmaxf(mx, __shfl_xor(mx, 2, 32));
      mx = fmaxf(mx, __shfl_xor(mx, 4, 32));
      mx = fmaxf(mx, __shfl_xor(mx, 8, 32));
      mnew[r] = fmaxf(rm[r], mx);
      corr[r] = exp2f((rm[r] - mnew[r]) * L2E);
      rm[r]   = mnew[r];
    }
    float lsum[8];
#pragma unroll
    for (int r = 0; r < 8; ++r) lsum[r] = 0.0f;
#pragma unroll
    for (int j = 0; j < 4; ++j)
#pragma unroll
      for (int r = 0; r < 8; ++r) {
        float p = exp2f((sc[j][r] - mnew[r]) * L2E);
        sc[j][r] = p;
        lsum[r] += p;
      }
#pragma unroll
    for (int r = 0; r < 8; ++r) {
      float ls = lsum[r];
      ls += __shfl_xor(ls, 1, 32);
      ls += __shfl_xor(ls, 2, 32);
      ls += __shfl_xor(ls, 4, 32);
      ls += __shfl_xor(ls, 8, 32);
      rl[r] = rl[r] * corr[r] + ls;
#pragma unroll
      for (int j = 0; j < 4; ++j) o[j][r] *= corr[r];
    }

    // C-layout -> A-layout for P via per-wave LDS scratch
    _Float16* Pw = &Pl[wid][0];
    {
      const int rofs = (lane >> 4) << 3;
      const int cofs = lane & 15;
#pragma unroll
      for (int j = 0; j < 4; ++j)
#pragma unroll
        for (int r = 0; r < 8; ++r)
          Pw[(r + rofs) * 72 + j * 16 + cofs] = (_Float16)sc[j][r];
    }
    // Same-wave DS in-order + explicit drain before re-reading P
    asm volatile("s_wait_dscnt 0" ::: "memory");
    v16h aP0 = frag_a(Pw,      72, lane);   // keys 0..31
    v16h aP1 = frag_a(Pw + 32, 72, lane);   // keys 32..63
#pragma unroll
    for (int j = 0; j < 4; ++j) {
      v16h b0 = frag_bt(Vt + (j * 16) * 80,      80, lane);  // keys 0..31
      v16h b1 = frag_bt(Vt + (j * 16) * 80 + 32, 80, lane);  // keys 32..63
      o[j] = wmma32(aP0, b0, o[j]);
      o[j] = wmma32(aP1, b1, o[j]);
    }
  }

  // Normalize and store att in [B,N,H,HD] (row-major [B*N, E] for proj GEMM)
  const int rofs = (lane >> 4) << 3;
  const int cofs = lane & 15;
  const int b = bh / NH, h = bh % NH;
#pragma unroll
  for (int r = 0; r < 8; ++r) {
    float inv = 1.0f / rl[r];
    int n = q0 + wid * 16 + r + rofs;
#pragma unroll
    for (int j = 0; j < 4; ++j) {
      int d = j * 16 + cofs;
      att[(((size_t)b * SEQ + n) * NH + h) * HD + d] = (_Float16)(o[j][r] * inv);
    }
  }
}

// ============================================================================
// Kernel 3: out = att @ w_proj (f16 A, f32 W -> f16, f32 out).
// ============================================================================
__global__ __launch_bounds__(256) void proj_gemm_kernel(
    const _Float16* __restrict__ a, const float* __restrict__ w,
    float* __restrict__ out) {
  __shared__ _Float16 As[128 * 40];
  __shared__ _Float16 Bt[64 * 40];
  const int t    = threadIdx.x;
  const int lane = t & 31;
  const int wid  = t >> 5;
  const int wm   = wid & 3, wn = wid >> 2;
  const int mBase = blockIdx.y * 128;
  const int nBase = blockIdx.x * 64;

  v8f acc[2][2];
#pragma unroll
  for (int i = 0; i < 2; ++i)
#pragma unroll
    for (int j = 0; j < 2; ++j) acc[i][j] = {};

  for (int k0 = 0; k0 < EMB; k0 += 32) {
    __syncthreads();
#pragma unroll
    for (int i = 0; i < 2; ++i) {          // A already f16: 512 uint4 copies
      int idx = t + i * 256;
      int r = idx >> 2, c8 = (idx & 3) * 8;
      *(uint4*)&As[r * 40 + c8] =
          *(const uint4*)&a[(size_t)(mBase + r) * EMB + k0 + c8];
    }
#pragma unroll
    for (int i = 0; i < 2; ++i) {          // B: 32x64 fp32 = 512 float4
      int idx = t + i * 256;
      int k = idx >> 4, n4 = (idx & 15) * 4;
      float4 f = *(const float4*)&w[(size_t)(k0 + k) * EMB + nBase + n4];
      Bt[(n4 + 0) * 40 + k] = (_Float16)f.x;
      Bt[(n4 + 1) * 40 + k] = (_Float16)f.y;
      Bt[(n4 + 2) * 40 + k] = (_Float16)f.z;
      Bt[(n4 + 3) * 40 + k] = (_Float16)f.w;
    }
    if (k0 + 32 < EMB) {
      __builtin_prefetch(&a[(size_t)(mBase + (t >> 1)) * EMB + k0 + 32], 0, 1);
      if (t < 32)
        __builtin_prefetch(&w[(size_t)(k0 + 32 + t) * EMB + nBase], 0, 1);
    }
    __syncthreads();

    v16h a0 = frag_a(As + (wm * 32)      * 40, 40, lane);
    v16h a1 = frag_a(As + (wm * 32 + 16) * 40, 40, lane);
    v16h b0 = frag_bt(Bt + (wn * 32)      * 40, 40, lane);
    v16h b1 = frag_bt(Bt + (wn * 32 + 16) * 40, 40, lane);
    acc[0][0] = wmma32(a0, b0, acc[0][0]);
    acc[0][1] = wmma32(a0, b1, acc[0][1]);
    acc[1][0] = wmma32(a1, b0, acc[1][0]);
    acc[1][1] = wmma32(a1, b1, acc[1][1]);
  }

  const int rofs = (lane >> 4) << 3;
  const int cofs = lane & 15;
#pragma unroll
  for (int i = 0; i < 2; ++i)
#pragma unroll
    for (int j = 0; j < 2; ++j)
#pragma unroll
      for (int r = 0; r < 8; ++r) {
        int grow = mBase + wm * 32 + i * 16 + r + rofs;
        int gcol = nBase + wn * 32 + j * 16 + cofs;
        out[(size_t)grow * EMB + gcol] = acc[i][j][r];
      }
}

extern "C" void kernel_launch(void* const* d_in, const int* in_sizes, int n_in,
                              void* d_out, int out_size, void* d_ws,
                              size_t ws_size, hipStream_t stream) {
  (void)in_sizes; (void)n_in; (void)out_size; (void)ws_size;
  const float* x      = (const float*)d_in[0];
  const float* w_qkv  = (const float*)d_in[1];
  const float* w_proj = (const float*)d_in[2];
  float* out = (float*)d_out;

  // Workspace: Q,K,V [B,H,N,HD] f16 + att [B,N,H,HD] f16 = 4 x 12.6 MB
  _Float16* ws = (_Float16*)d_ws;
  const size_t per = (size_t)BQ * NH * SEQ * HD;
  _Float16* qb  = ws;
  _Float16* kb  = ws + per;
  _Float16* vb  = ws + 2 * per;
  _Float16* att = ws + 3 * per;

  qkv_gemm_kernel<<<dim3(QKVN / 64, (BQ * SEQ) / 128), 256, 0, stream>>>(
      x, w_qkv, qb, kb, vb);
  attn_kernel<<<dim3(SEQ / 64, BQ * NH), 128, 0, stream>>>(qb, kb, vb, att);
  proj_gemm_kernel<<<dim3(EMB / 64, (BQ * SEQ) / 128), 256, 0, stream>>>(
      att, w_proj, out);
}
